// MultiheadLatentAttention_2911987827441
// MI455X (gfx1250) — matrix operand order repaired and checked by reading, not compiled, not run on hardware
//
#include <hip/hip_runtime.h>
#include <hip/hip_bf16.h>
#include <math.h>

// ---------------------------------------------------------------------------
// MLA forward for MI455X (gfx1250, wave32, WMMA).
//   latent:(2,1024,1024) x:(2,4096,2048)
//   Q=latent@Wq+bq ; C=x@Wc+bc ; K=C@Wk+bk ; V=C@Wv+bv
//   P=softmax(Q K^T / 8) ; O=(P V) @ Wo + bo
// d_out = [output (2*1024*1024 f32)] ++ [attn_weights (2*16*1024*4096 f32)]
// ---------------------------------------------------------------------------

typedef __attribute__((ext_vector_type(16))) __bf16 v16bf;
typedef __attribute__((ext_vector_type(8)))  float  v8f;

#define B_DIM   2
#define L_DIM   1024
#define S_DIM   4096
#define H_NUM   16
#define D_HEAD  64
#define MODEL   1024
#define IN_DIM  2048
#define RANK    256

// native f32 -> bf16 conversion (gfx1250 has hardware bf16 cvt)
__device__ __forceinline__ __bf16 f2bf(float f) { return (__bf16)f; }

// ---- WMMA fragment loaders (bf16 16x16x32, contiguous k = 16*half + s, wave32) ----------
// A 16x32 (MxK): lanes 0-15 -> M, half selects K sub-block; VGPR v pairs K.
__device__ __forceinline__ v16bf load_a_f32(const float* A, int row0, int k0, int lda) {
  int lane = threadIdx.x & 31;
  int m = lane & 15, hl = lane >> 4;
  const float* base = A + (size_t)(row0 + m) * lda + k0;
  v16bf a;
#pragma unroll
  for (int v = 0; v < 8; ++v) {
    int k = hl * 16 + 2 * v;
    a[2 * v]     = f2bf(base[k]);
    a[2 * v + 1] = f2bf(base[k + 1]);
  }
  return a;
}

__device__ __forceinline__ v16bf load_a_bf16(const __bf16* A, int row0, int k0, int lda) {
  int lane = threadIdx.x & 31;
  int m = lane & 15, hl = lane >> 4;
  const __bf16* base = A + (size_t)(row0 + m) * lda + k0;
  v16bf a;
#pragma unroll
  for (int v = 0; v < 8; ++v) {
    int k = hl * 16 + 2 * v;
    a[2 * v]     = base[k];
    a[2 * v + 1] = base[k + 1];
  }
  return a;
}

// B 32x16 (KxN) from row-major src: element(k,n) = B[(k0+k)*ldb + n0+n]
__device__ __forceinline__ v16bf load_b_f32(const float* B, int k0, int n0, int ldb) {
  int lane = threadIdx.x & 31;
  int n = lane & 15, hl = lane >> 4;
  const float* base = B + (size_t)(k0 + hl * 16) * ldb + n0 + n;
  v16bf b;
#pragma unroll
  for (int v = 0; v < 8; ++v) {
    b[2 * v]     = f2bf(base[(size_t)(2 * v) * ldb]);
    b[2 * v + 1] = f2bf(base[(size_t)(2 * v + 1) * ldb]);
  }
  return b;
}

__device__ __forceinline__ v16bf load_b_bf16(const __bf16* B, size_t k0row, int n0, int ldb) {
  int lane = threadIdx.x & 31;
  int n = lane & 15, hl = lane >> 4;
  const __bf16* base = B + (k0row + hl * 16) * (size_t)ldb + n0 + n;
  v16bf b;
#pragma unroll
  for (int v = 0; v < 8; ++v) {
    b[2 * v]     = base[(size_t)(2 * v) * ldb];
    b[2 * v + 1] = base[(size_t)(2 * v + 1) * ldb];
  }
  return b;
}

// B^T fragment for Q*K^T: element(k,n) = S[(row0+n)*ld + col0+k]  (contiguous k)
__device__ __forceinline__ v16bf load_bt_bf16(const __bf16* S, size_t row0, int col0, int ld) {
  int lane = threadIdx.x & 31;
  int n = lane & 15, hl = lane >> 4;
  const __bf16* base = S + (row0 + n) * (size_t)ld + col0 + hl * 16;
  v16bf b;
#pragma unroll
  for (int v = 0; v < 8; ++v) {
    b[2 * v]     = base[2 * v];
    b[2 * v + 1] = base[2 * v + 1];
  }
  return b;
}

// ---------------------------------------------------------------------------
// Generic bias GEMM: out[M,N] = A[M,K] (f32) * B[K,N] (f32) + bias[N].
// One 16x16 tile per wave, 8 waves per block, bf16 WMMA / f32 accumulate.
// Output precision is a compile-time choice -> straight-line store epilogue.
// ---------------------------------------------------------------------------
template <bool BF16OUT>
__global__ void __launch_bounds__(256)
wmma_gemm_bias(const float* __restrict__ A, const float* __restrict__ B,
               const float* __restrict__ bias, float* __restrict__ outF,
               __bf16* __restrict__ outB, int M, int N, int K) {
  int wave = threadIdx.x >> 5;
  int tid = blockIdx.x * 8 + wave;
  int ntiles = N >> 4;
  int total = (M >> 4) * ntiles;
  if (tid >= total) return;                       // wave-uniform
  int row0 = (tid / ntiles) << 4;
  int col0 = (tid % ntiles) << 4;
  int lane = threadIdx.x & 31;
  int hl = lane >> 4, n = lane & 15;

  v8f acc = {};
  for (int k0 = 0; k0 < K; k0 += 32) {
    if (k0 + 32 < K) {                            // warm WGP$ for next K step
      __builtin_prefetch(A + (size_t)(row0 + (lane & 15)) * K + k0 + 32, 0, 1);
      __builtin_prefetch(B + (size_t)(k0 + 32 + lane) * N + col0, 0, 1);
    }
    v16bf a = load_a_f32(A, row0, k0, K);
    v16bf b = load_b_f32(B, k0, col0, N);
    acc = __builtin_amdgcn_wmma_f32_16x16x32_bf16(false, a, false, b,
                                                  (short)0, acc, false, false);
  }

  float bv = bias[col0 + n];
#pragma unroll
  for (int r = 0; r < 8; ++r) {
    int row = row0 + r + hl * 8;                  // C/D layout: VGPR r -> M=r(+8)
    size_t idx = (size_t)row * N + col0 + n;
    float v = acc[r] + bv;
    if constexpr (BF16OUT) outB[idx] = f2bf(v);
    else                   outF[idx] = v;
  }
}

// ---------------------------------------------------------------------------
// Attention: one block per (b, h, 16-row L tile). Full 16x4096 f32 score panel
// lives in LDS (256 KB of the WGP's 320 KB): WMMA scores -> in-place softmax ->
// stream normalized P to d_out -> WMMA P@V consumed straight from LDS.
// ---------------------------------------------------------------------------
__global__ void __launch_bounds__(256)
mla_attention(const __bf16* __restrict__ Qbf, const __bf16* __restrict__ Kbf,
              const __bf16* __restrict__ Vbf, float* __restrict__ attnW,
              float* __restrict__ attnO) {
  extern __shared__ float smem[];                 // 16 * 4096 f32 = 256 KB
  const int lt   = blockIdx.x;                    // 0..63  (L tile)
  const int h    = blockIdx.y;                    // 0..15
  const int b    = blockIdx.z;                    // 0..1
  const int wave = threadIdx.x >> 5;
  const int lane = threadIdx.x & 31;
  const int hl   = lane >> 4, nn = lane & 15;
  const int l0   = lt * 16;
  const int qrow0 = b * L_DIM + l0;
  const int dcol  = h * D_HEAD;

  // ---- scores = Q K^T / sqrt(D) into LDS (S tiles round-robined over waves)
  v16bf qa0 = load_a_bf16(Qbf, qrow0, dcol,      MODEL);
  v16bf qa1 = load_a_bf16(Qbf, qrow0, dcol + 32, MODEL);
  for (int t = wave; t < S_DIM / 16; t += 8) {
    int s0 = t * 16;
    size_t krow = (size_t)b * S_DIM + s0;
    v16bf kb0 = load_bt_bf16(Kbf, krow, dcol,      MODEL);
    v16bf kb1 = load_bt_bf16(Kbf, krow, dcol + 32, MODEL);
    v8f acc = {};
    acc = __builtin_amdgcn_wmma_f32_16x16x32_bf16(false, qa0, false, kb0,
                                                  (short)0, acc, false, false);
    acc = __builtin_amdgcn_wmma_f32_16x16x32_bf16(false, qa1, false, kb1,
                                                  (short)0, acc, false, false);
#pragma unroll
    for (int r = 0; r < 8; ++r)
      smem[(r + hl * 8) * S_DIM + s0 + nn] = acc[r] * 0.125f;  // 1/sqrt(64)
  }
  __syncthreads();

  // ---- softmax per row (16 lanes per row), write normalized P to HBM + LDS
  {
    int r = threadIdx.x >> 4;                     // row 0..15
    int j = threadIdx.x & 15;
    float* row = smem + r * S_DIM;
    float mx = -INFINITY;
    for (int s = j; s < S_DIM; s += 16) mx = fmaxf(mx, row[s]);
#pragma unroll
    for (int off = 8; off; off >>= 1) mx = fmaxf(mx, __shfl_xor(mx, off));
    float sum = 0.f;
    for (int s = j; s < S_DIM; s += 16) {
      float e = __expf(row[s] - mx);
      row[s] = e;
      sum += e;
    }
#pragma unroll
    for (int off = 8; off; off >>= 1) sum += __shfl_xor(sum, off);
    float inv = 1.f / sum;
    float* wrow = attnW + ((size_t)(b * H_NUM + h) * L_DIM + l0 + r) * S_DIM;
    for (int s = j; s < S_DIM; s += 16) {
      float p = row[s] * inv;
      row[s] = p;
      wrow[s] = p;                                // the one mandatory HBM stream
    }
  }
  __syncthreads();

  // ---- attn_out tile (16 x 64) = P @ V ; waves split (D slice, S half)
  int ntile = wave & 3;                           // 16-wide D slice
  int khalf = wave >> 2;                          // 0/1: S half
  int vcol = dcol + ntile * 16;
  v8f acc = {};
  for (int s0 = khalf * (S_DIM / 2); s0 < (khalf + 1) * (S_DIM / 2); s0 += 32) {
    v16bf pa = load_a_f32(smem, 0, s0, S_DIM);    // P from LDS
    v16bf vb = load_b_bf16(Vbf, (size_t)b * S_DIM + s0, vcol, MODEL);
    acc = __builtin_amdgcn_wmma_f32_16x16x32_bf16(false, pa, false, vb,
                                                  (short)0, acc, false, false);
  }
  __syncthreads();                                // done reading P; reuse LDS
  float* red = smem;
  if (khalf == 1) {
#pragma unroll
    for (int r = 0; r < 8; ++r) red[(ntile * 32 + lane) * 8 + r] = acc[r];
  }
  __syncthreads();
  if (khalf == 0) {
#pragma unroll
    for (int r = 0; r < 8; ++r) {
      float v = acc[r] + red[(ntile * 32 + lane) * 8 + r];
      int row = l0 + r + hl * 8;
      attnO[((size_t)b * L_DIM + row) * MODEL + vcol + nn] = v;
    }
  }
}

// ---------------------------------------------------------------------------
extern "C" void kernel_launch(void* const* d_in, const int* in_sizes, int n_in,
                              void* d_out, int out_size, void* d_ws, size_t ws_size,
                              hipStream_t stream) {
  const float* latent = (const float*)d_in[0];
  const float* x      = (const float*)d_in[1];
  const float* Wq = (const float*)d_in[2];  const float* bq = (const float*)d_in[3];
  const float* Wc = (const float*)d_in[4];  const float* bc = (const float*)d_in[5];
  const float* Wk = (const float*)d_in[6];  const float* bk = (const float*)d_in[7];
  const float* Wv = (const float*)d_in[8];  const float* bv = (const float*)d_in[9];
  const float* Wo = (const float*)d_in[10]; const float* bo = (const float*)d_in[11];

  float* out   = (float*)d_out;                            // (B,L,MODEL)
  float* attnW = out + (size_t)B_DIM * L_DIM * MODEL;      // (B,H,L,S)

  // workspace carve-up (~52 MB)
  char* ws = (char*)d_ws;
  __bf16* Qbf = (__bf16*)ws;  ws += (size_t)B_DIM * L_DIM * MODEL * sizeof(__bf16);
  __bf16* Kbf = (__bf16*)ws;  ws += (size_t)B_DIM * S_DIM * MODEL * sizeof(__bf16);
  __bf16* Vbf = (__bf16*)ws;  ws += (size_t)B_DIM * S_DIM * MODEL * sizeof(__bf16);
  float*  Cf  = (float*)ws;   ws += (size_t)B_DIM * S_DIM * RANK  * sizeof(float);
  float*  AO  = (float*)ws;   ws += (size_t)B_DIM * L_DIM * MODEL * sizeof(float);

  auto blocks = [](int M, int N) { return ((M >> 4) * (N >> 4) + 7) / 8; };

  // 1) Q = latent @ Wq + bq        -> bf16 (2048 x 1024)
  wmma_gemm_bias<true><<<blocks(B_DIM * L_DIM, MODEL), 256, 0, stream>>>(
      latent, Wq, bq, nullptr, Qbf, B_DIM * L_DIM, MODEL, MODEL);
  // 2) C = x @ Wc + bc             -> f32  (8192 x 256)
  wmma_gemm_bias<false><<<blocks(B_DIM * S_DIM, RANK), 256, 0, stream>>>(
      x, Wc, bc, Cf, nullptr, B_DIM * S_DIM, RANK, IN_DIM);
  // 3) K = C @ Wk + bk             -> bf16 (8192 x 1024)
  wmma_gemm_bias<true><<<blocks(B_DIM * S_DIM, MODEL), 256, 0, stream>>>(
      Cf, Wk, bk, nullptr, Kbf, B_DIM * S_DIM, MODEL, RANK);
  // 4) V = C @ Wv + bv             -> bf16 (8192 x 1024)
  wmma_gemm_bias<true><<<blocks(B_DIM * S_DIM, MODEL), 256, 0, stream>>>(
      Cf, Wv, bv, nullptr, Vbf, B_DIM * S_DIM, MODEL, RANK);
  // 5) attention: scores->softmax->P@V, 256 KB LDS score panel per block
  dim3 agrid(L_DIM / 16, H_NUM, B_DIM);
  mla_attention<<<agrid, 256, 16 * S_DIM * sizeof(float), stream>>>(
      Qbf, Kbf, Vbf, attnW, AO);
  // 6) output = attn_out @ Wo + bo -> f32 into d_out
  wmma_gemm_bias<false><<<blocks(B_DIM * L_DIM, MODEL), 256, 0, stream>>>(
      AO, Wo, bo, out, nullptr, B_DIM * L_DIM, MODEL, MODEL);
}
